// GlobalAttentionPooling_42958262894957
// MI455X (gfx1250) — matrix operand order repaired and checked by reading, not compiled, not run on hardware
//
#include <hip/hip_runtime.h>
#include <math.h>

// CDNA5 wave32 WMMA vector types
typedef __attribute__((ext_vector_type(16))) _Float16 v16h;
typedef __attribute__((ext_vector_type(8)))  float    v8f;

#define DDIM 128
#define CHUNK 128

// Monotone float<->uint mapping so atomicMax(uint) implements float max.
__device__ __forceinline__ unsigned f32_ord(float f) {
  unsigned u = __float_as_uint(f);
  return (u & 0x80000000u) ? ~u : (u | 0x80000000u);
}
__device__ __forceinline__ float ord_f32(unsigned o) {
  return (o & 0x80000000u) ? __uint_as_float(o & 0x7FFFFFFFu)
                           : __uint_as_float(~o);
}

// ---------------------------------------------------------------- init
__global__ void k_init(float* __restrict__ out, unsigned* __restrict__ segmax,
                       float* __restrict__ segsum, int nOut, int nSeg) {
  int i = blockIdx.x * blockDim.x + threadIdx.x;
  if (i < nOut) out[i] = 0.0f;
  if (i < nSeg) { segmax[i] = 0u; segsum[i] = 0.0f; }
}

// ------------------------------------------------- gate = feat @ W + b (WMMA)
// One wave handles a 16-node tile. A = feat tile (16x32 f16 per step, 4 K-steps),
// B = W_gate broadcast into all 16 columns (constant across tiles).
// D[m][n] == gate[m] for every n; lanes 0/16 extract and store.
__global__ void k_gate_wmma(const float* __restrict__ feat,
                            const float* __restrict__ Wg,
                            const float* __restrict__ bgate,
                            const int*   __restrict__ sid,
                            float*       __restrict__ gate,
                            unsigned*    __restrict__ segmax,
                            int ntiles, int n) {
  const int lane = threadIdx.x & 31;
  const int hi   = lane >> 4;      // half-wave select
  const int r    = lane & 15;
  const int wave  = (blockIdx.x * blockDim.x + threadIdx.x) >> 5;
  const int nwave = (gridDim.x * blockDim.x) >> 5;

  // B-matrix 32x16 f16 layout: lane holds K = kc+8*hi+[0..7] in v0..v3 and
  // K = kc+16+8*hi+[0..7] in v4..v7 (col = lane). Broadcast W into every column.
  v16h bmat;
  const float bias = bgate[0];

  for (int t = wave; t < ntiles; t += nwave) {
    int row = t * 16 + r;
    if (row >= n) row = n - 1;                       // clamp (loads only)
    const float* frow = feat + (size_t)row * DDIM;
    v8f c = {};
#pragma unroll
    for (int kc = 0; kc < DDIM; kc += 32) {
      // A-matrix 16x32 f16 layout: lane(=row) holds K = kc+8*hi+[0..7]
      // in v0..v3 and K = kc+16+8*hi+[0..7] in v4..v7.
      v16h a;
#pragma unroll
      for (int j = 0; j < 8; ++j) {
        a[j]     = (_Float16)frow[kc + 8 * hi + j];
        a[8 + j] = (_Float16)frow[kc + 16 + 8 * hi + j];
        bmat[j]     = (_Float16)Wg[kc + 8 * hi + j];
        bmat[8 + j] = (_Float16)Wg[kc + 16 + 8 * hi + j];
      }
      c = __builtin_amdgcn_wmma_f32_16x16x32_f16(false, a, false, bmat,
                                                 (short)0, c, false, false);
    }
    // D layout: VGPR m, lanes 0-15 -> M=m, lanes 16-31 -> M=m+8.
    // All columns identical, so lane 0 has gate[tile..tile+7],
    // lane 16 has gate[tile+8..tile+15].
    if (r == 0) {
#pragma unroll
      for (int m = 0; m < 8; ++m) {
        int node = t * 16 + 8 * hi + m;
        if (node < n) {
          float g = c[m] + bias;
          gate[node] = g;
          atomicMax(&segmax[sid[node]], f32_ord(g));
        }
      }
    }
  }
}

// ---------------------------------------------------------------- exp + sum
__global__ void k_exp(const int* __restrict__ sid,
                      const unsigned* __restrict__ segmax,
                      float* __restrict__ gate,          // in: gate, out: e
                      float* __restrict__ segsum, int n) {
  int i = blockIdx.x * blockDim.x + threadIdx.x;
  if (i >= n) return;
  int s = sid[i];
  float e = __expf(gate[i] - ord_f32(segmax[s]));
  gate[i] = e;
  atomicAdd(&segsum[s], e);
}

// ---------------------------------------------------------------- 1/segsum
__global__ void k_inv(const float* __restrict__ segsum,
                      float* __restrict__ inv, int nseg) {
  int i = blockIdx.x * blockDim.x + threadIdx.x;
  if (i < nseg) inv[i] = 1.0f / segsum[i];   // empty segment -> inf, never read
}

// ------------------------------------------------------- gated segment sum
// One wave per CHUNK contiguous nodes; lane owns 4 feature dims (float4 loads).
// segment_ids are sorted, so accumulate runs in registers and flush with
// atomicAdd only at segment boundaries / chunk end.
__global__ void k_readout(const float* __restrict__ feat,
                          const float* __restrict__ e,
                          const int*   __restrict__ sid,
                          const float* __restrict__ inv,
                          float*       __restrict__ out, int n) {
  const int lane  = threadIdx.x & 31;
  const int chunk = blockIdx.x * (blockDim.x >> 5) + (threadIdx.x >> 5);
  int start = chunk * CHUNK;
  if (start >= n) return;
  int end = start + CHUNK;
  if (end > n) end = n;

  const int d4 = lane * 4;
  float4 acc = make_float4(0.f, 0.f, 0.f, 0.f);
  int cur = sid[start];

  for (int i = start; i < end; ++i) {
    int s = sid[i];                       // wave-uniform
    if (s != cur) {
      float* o = out + (size_t)cur * DDIM + d4;
      atomicAdd(o + 0, acc.x);
      atomicAdd(o + 1, acc.y);
      atomicAdd(o + 2, acc.z);
      atomicAdd(o + 3, acc.w);
      acc = make_float4(0.f, 0.f, 0.f, 0.f);
      cur = s;
    }
    float a = e[i] * inv[s];
    const float4 f = *(const float4*)(feat + (size_t)i * DDIM + d4);
    acc.x += f.x * a;
    acc.y += f.y * a;
    acc.z += f.z * a;
    acc.w += f.w * a;
  }
  float* o = out + (size_t)cur * DDIM + d4;
  atomicAdd(o + 0, acc.x);
  atomicAdd(o + 1, acc.y);
  atomicAdd(o + 2, acc.z);
  atomicAdd(o + 3, acc.w);
}

// ---------------------------------------------------------------- launcher
extern "C" void kernel_launch(void* const* d_in, const int* in_sizes, int n_in,
                              void* d_out, int out_size, void* d_ws, size_t ws_size,
                              hipStream_t stream) {
  const float* feat  = (const float*)d_in[0];
  const float* Wg    = (const float*)d_in[1];
  const float* bgate = (const float*)d_in[2];
  const int*   sid   = (const int*)d_in[3];

  const int n    = in_sizes[3];          // N nodes
  const int nseg = out_size / DDIM;      // B segments

  // Workspace carve-up (all 4-byte aligned)
  char* ws = (char*)d_ws;
  float*    gate   = (float*)ws;                       // N floats (gate -> e)
  unsigned* segmax = (unsigned*)(ws + (size_t)n * 4);  // B uints
  float*    segsum = (float*)((char*)segmax + (size_t)nseg * 4);
  float*    segInv = (float*)((char*)segsum + (size_t)nseg * 4);

  // 1. init out / segmax / segsum
  {
    int tot = out_size > nseg ? out_size : nseg;
    int blocks = (tot + 255) / 256;
    k_init<<<blocks, 256, 0, stream>>>((float*)d_out, segmax, segsum, out_size, nseg);
  }
  // 2. gate projection via WMMA + segment max
  {
    int ntiles = (n + 15) / 16;
    int blocks = 2048;
    int maxBlocks = (ntiles + 7) / 8;    // 8 waves per 256-thread block
    if (blocks > maxBlocks) blocks = maxBlocks;
    if (blocks < 1) blocks = 1;
    k_gate_wmma<<<blocks, 256, 0, stream>>>(feat, Wg, bgate, sid, gate, segmax,
                                            ntiles, n);
  }
  // 3. stable exp + segment sum
  k_exp<<<(n + 255) / 256, 256, 0, stream>>>(sid, segmax, gate, segsum, n);
  // 4. reciprocal of segment sums
  k_inv<<<(nseg + 255) / 256, 256, 0, stream>>>(segsum, segInv, nseg);
  // 5. gated segmented readout
  {
    int chunks = (n + CHUNK - 1) / CHUNK;
    int blocks = (chunks + 3) / 4;       // 4 waves (chunks) per 128-thread block
    k_readout<<<blocks, 128, 0, stream>>>(feat, gate, sid, segInv,
                                          (float*)d_out, n);
  }
}
